// ContextGNNLayer_90305982365987
// MI455X (gfx1250) — compile-verified, matrix operand order checked by reading, not run on hardware
//
#include <hip/hip_runtime.h>

// Problem constants (match reference)
#define Nn 100000
#define Ee 1600000
#define Dd 128
#define Hh 8
#define Cc 16

typedef __attribute__((ext_vector_type(16))) __bf16 v16bf;
typedef __attribute__((ext_vector_type(8)))  float v8f;

#if __has_builtin(__builtin_amdgcn_global_load_async_to_lds_b128)
#define HAS_ASYNC_LDS 1
// Pointer types matching the builtin's signature (from the round-2 diagnostic):
// param0: vector-of-4-int pointer in global (device) address space.
typedef int v4i_vs __attribute__((vector_size(16)));
typedef __attribute__((address_space(1))) v4i_vs* gv4i_p;
typedef __attribute__((address_space(3))) v4i_vs* lv4i_p;
#else
#define HAS_ASYNC_LDS 0
#endif

__device__ __forceinline__ unsigned short f2bf(float f) {
  unsigned int u = __float_as_uint(f);
  u += 0x7FFFu + ((u >> 16) & 1u);   // round-to-nearest-even
  return (unsigned short)(u >> 16);
}

// Deterministic float atomic max via int/uint ordering trick (init = -inf)
__device__ __forceinline__ void atomicMaxF(float* addr, float v) {
  if (v >= 0.0f) atomicMax((int*)addr, __float_as_int(v));
  else           atomicMin((unsigned int*)addr, __float_as_uint(v));
}

// ---- init: zero accumulator (d_out), m = -inf, s = 0 ----
__global__ void k_init(float* __restrict__ out, float* __restrict__ m,
                       float* __restrict__ s) {
  int i = blockIdx.x * blockDim.x + threadIdx.x;
  if (i < Nn * Dd) out[i] = 0.0f;
  if (i < Nn * Hh) { m[i] = __uint_as_float(0xFF800000u); s[i] = 0.0f; }
}

// ---- transpose + convert W[k][n] f32 -> Wt[n][k] bf16 ----
__global__ void k_wcvt(const float* __restrict__ W, unsigned short* __restrict__ Wt) {
  int t = blockIdx.x * blockDim.x + threadIdx.x;
  if (t >= Dd * Dd) return;
  int n = t >> 7, k = t & 127;
  Wt[n * Dd + k] = f2bf(W[k * Dd + n]);
}

// ---- GEMM: h = x @ W via v_wmma_f32_16x16x32_bf16 ----
#define TM   64     // rows per block
#define LDSW 136    // padded LDS row stride (ushorts): 272B, 16B-aligned rows
__global__ void __launch_bounds__(256) k_gemm(const float* __restrict__ x,
                                              const unsigned short* __restrict__ Wt,
                                              float* __restrict__ h) {
  __shared__ unsigned short Xl[TM * LDSW];
  __shared__ unsigned short Wl[Dd * LDSW];
  const int t = threadIdx.x;
  const int R = blockIdx.x * TM;

  // W tile (128x128 bf16 = 2048 x 16B): async global->LDS copy (ASYNCcnt path),
  // overlaps with the VALU-side X conversion below.
#if HAS_ASYNC_LDS
  #pragma unroll
  for (int i = 0; i < 8; ++i) {
    int q = i * 256 + t;
    int row = q >> 4, c8 = (q & 15) << 3;
    __builtin_amdgcn_global_load_async_to_lds_b128(
        (gv4i_p)(Wt + q * 8),
        (lv4i_p)&Wl[row * LDSW + c8], 0, 0);
  }
#else
  #pragma unroll
  for (int i = 0; i < 8; ++i) {
    int q = i * 256 + t;
    int row = q >> 4, c8 = (q & 15) << 3;
    *(uint4*)&Wl[row * LDSW + c8] = ((const uint4*)Wt)[q];
  }
#endif

  // X tile (64x128 f32), convert to bf16 through VGPRs
  #pragma unroll
  for (int i = 0; i < 8; ++i) {
    int q = i * 256 + t;
    int row = q >> 5, c4 = (q & 31) << 2;
    int gr = R + row;
    float4 xv = make_float4(0.f, 0.f, 0.f, 0.f);
    if (gr < Nn) xv = *(const float4*)(x + (size_t)gr * Dd + c4);
    unsigned int u0 = (unsigned int)f2bf(xv.x) | ((unsigned int)f2bf(xv.y) << 16);
    unsigned int u1 = (unsigned int)f2bf(xv.z) | ((unsigned int)f2bf(xv.w) << 16);
    *(uint2*)&Xl[row * LDSW + c4] = make_uint2(u0, u1);
  }

#if HAS_ASYNC_LDS
#if __has_builtin(__builtin_amdgcn_s_wait_asynccnt)
  __builtin_amdgcn_s_wait_asynccnt(0);
#else
  asm volatile("s_wait_asynccnt 0x0" ::: "memory");
#endif
#endif
  __syncthreads();

  const int lane = t & 31;
  const int w    = t >> 5;        // 8 waves
  const int r    = lane & 15;
  const int hi   = lane >> 4;     // lane half selects K sub-range / row+8
  const int m0   = (w & 3) * 16;  // 4 row tiles
  const bool full = (R + TM) <= Nn;   // block-uniform bounds check

  union FragU { v16bf v; uint4 q[2]; };

  for (int j = 0; j < 4; ++j) {
    const int ct = (w >> 2) * 4 + j;  // 8 col tiles split across wave pairs
    const int n0 = ct * 16;
    v8f acc = {};
    #pragma unroll
    for (int kb = 0; kb < Dd; kb += 32) {
      FragU a, b;
      // A: 16x32 bf16 — lane r = row, halves carry K {0..7,16..23} / {8..15,24..31}
      const unsigned short* ap = &Xl[(m0 + r) * LDSW + kb + hi * 8];
      a.q[0] = *(const uint4*)ap;
      a.q[1] = *(const uint4*)(ap + 16);
      // B loaded from W^T with identical layout (lane r = output col)
      const unsigned short* bp = &Wl[(n0 + r) * LDSW + kb + hi * 8];
      b.q[0] = *(const uint4*)bp;
      b.q[1] = *(const uint4*)(bp + 16);
      acc = __builtin_amdgcn_wmma_f32_16x16x32_bf16(false, a.v, false, b.v,
                                                    (short)0, acc, false, false);
    }
    // D layout: VGPR v -> row m0 + v + 8*hi, col n0 + r.
    // Base pointer + immediate offsets (v*512B); branch-free for full blocks.
    const int rbase = R + m0 + hi * 8;
    float* bp = h + (size_t)rbase * Dd + n0 + r;
    if (full) {
      #pragma unroll
      for (int v = 0; v < 8; ++v) bp[v * Dd] = acc[v];
    } else {
      #pragma unroll
      for (int v = 0; v < 8; ++v) if (rbase + v < Nn) bp[v * Dd] = acc[v];
    }
  }
}

// ---- per-node attention terms: a_src/a_dst [N,H] ----
__global__ void k_attn(const float* __restrict__ h, const float* __restrict__ asv,
                       const float* __restrict__ adv, float* __restrict__ a_src,
                       float* __restrict__ a_dst) {
  int t = blockIdx.x * blockDim.x + threadIdx.x;
  if (t >= Nn * Hh) return;
  int n = t >> 3, hh = t & 7;
  const float4* hp = (const float4*)(h + (size_t)n * Dd + hh * Cc);
  const float4* sp = (const float4*)(asv + hh * Cc);
  const float4* dp = (const float4*)(adv + hh * Cc);
  float s = 0.f, d = 0.f;
  #pragma unroll
  for (int i = 0; i < 4; ++i) {
    float4 hv = hp[i], sv = sp[i], dv = dp[i];
    s += hv.x * sv.x + hv.y * sv.y + hv.z * sv.z + hv.w * sv.w;
    d += hv.x * dv.x + hv.y * dv.y + hv.z * dv.z + hv.w * dv.w;
  }
  a_src[t] = s; a_dst[t] = d;
}

// Load an 8-float row as two float4 and produce leaky-relu'd per-head scores
__device__ __forceinline__ void edge_scores(const float* __restrict__ a_src,
                                            const float* __restrict__ a_dst,
                                            int src, int dst, float v[8]) {
  const float4* as = (const float4*)(a_src + src * Hh);
  const float4* ad = (const float4*)(a_dst + dst * Hh);
  float4 s0 = as[0], s1 = as[1], d0 = ad[0], d1 = ad[1];
  v[0] = s0.x + d0.x; v[1] = s0.y + d0.y; v[2] = s0.z + d0.z; v[3] = s0.w + d0.w;
  v[4] = s1.x + d1.x; v[5] = s1.y + d1.y; v[6] = s1.z + d1.z; v[7] = s1.w + d1.w;
  #pragma unroll
  for (int hh = 0; hh < Hh; ++hh) v[hh] = v[hh] >= 0.f ? v[hh] : 0.2f * v[hh];
}

// ---- edge pass 1: segment max ----
__global__ void k_emax(const int* __restrict__ ei, const float* __restrict__ a_src,
                       const float* __restrict__ a_dst, float* __restrict__ m) {
  int e = blockIdx.x * blockDim.x + threadIdx.x;
  if (e >= Ee) return;
  int src = ei[e], dst = ei[Ee + e];
  float v[8];
  edge_scores(a_src, a_dst, src, dst, v);
  #pragma unroll
  for (int hh = 0; hh < Hh; ++hh) atomicMaxF(&m[dst * Hh + hh], v[hh]);
}

// ---- edge pass 2: exp + segment sum (recompute e; avoids 51MB edge buffer) ----
__global__ void k_esum(const int* __restrict__ ei, const float* __restrict__ a_src,
                       const float* __restrict__ a_dst, const float* __restrict__ m,
                       float* __restrict__ s) {
  int e = blockIdx.x * blockDim.x + threadIdx.x;
  if (e >= Ee) return;
  int src = ei[e], dst = ei[Ee + e];
  float v[8];
  edge_scores(a_src, a_dst, src, dst, v);
  const float4* mp = (const float4*)(m + dst * Hh);
  float4 m0 = mp[0], m1 = mp[1];
  float mm[8] = {m0.x, m0.y, m0.z, m0.w, m1.x, m1.y, m1.z, m1.w};
  #pragma unroll
  for (int hh = 0; hh < Hh; ++hh)
    atomicAdd(&s[dst * Hh + hh], __expf(v[hh] - mm[hh]));
}

// ---- edge pass 3: weighted scatter, one wave (32 lanes) per edge ----
__global__ void __launch_bounds__(256) k_scatter(const int* __restrict__ ei,
    const float* __restrict__ a_src, const float* __restrict__ a_dst,
    const float* __restrict__ m, const float* __restrict__ s,
    const float* __restrict__ h, float* __restrict__ out) {
  int gid = blockIdx.x * blockDim.x + threadIdx.x;
  int e = gid >> 5;
  if (e >= Ee) return;
  int lane = gid & 31;           // lane covers cols 4*lane..4*lane+3, head = lane/4
  int src = ei[e], dst = ei[Ee + e];
  int head = lane >> 2;
  float v = a_src[src * Hh + head] + a_dst[dst * Hh + head];
  v = v >= 0.f ? v : 0.2f * v;
  float ex = __expf(v - m[dst * Hh + head]);
  float alpha = ex / (s[dst * Hh + head] + 1e-16f);
  float4 hv = *(const float4*)(h + (size_t)src * Dd + lane * 4);
  float* op = out + (size_t)dst * Dd + lane * 4;
  atomicAdd(op + 0, hv.x * alpha);
  atomicAdd(op + 1, hv.y * alpha);
  atomicAdd(op + 2, hv.z * alpha);
  atomicAdd(op + 3, hv.w * alpha);
}

// ---- finalize: +bias +residual, LayerNorm, ReLU. One wave per row. ----
__global__ void k_final(const float* __restrict__ x, const float* __restrict__ bias,
                        const float* __restrict__ gamma, const float* __restrict__ beta,
                        float* __restrict__ out) {
  int gid = blockIdx.x * blockDim.x + threadIdx.x;
  int row = gid >> 5;
  if (row >= Nn) return;
  int lane = gid & 31;
  int c = lane * 4;
  float4 o  = *(float4*)(out + (size_t)row * Dd + c);
  float4 xb = *(const float4*)(x + (size_t)row * Dd + c);
  float4 bb = *(const float4*)(bias + c);
  o.x += xb.x + bb.x; o.y += xb.y + bb.y; o.z += xb.z + bb.z; o.w += xb.w + bb.w;
  float sum = o.x + o.y + o.z + o.w;
  float sq  = o.x*o.x + o.y*o.y + o.z*o.z + o.w*o.w;
  #pragma unroll
  for (int off = 16; off > 0; off >>= 1) {  // wave32 tree reduction
    sum += __shfl_xor(sum, off, 32);
    sq  += __shfl_xor(sq,  off, 32);
  }
  float mean = sum * (1.0f / Dd);
  float var  = sq  * (1.0f / Dd) - mean * mean;
  float inv  = rsqrtf(var + 1e-5f);
  float4 g  = *(const float4*)(gamma + c);
  float4 bt = *(const float4*)(beta + c);
  float4 res;
  res.x = fmaxf((o.x - mean) * inv * g.x + bt.x, 0.f);
  res.y = fmaxf((o.y - mean) * inv * g.y + bt.y, 0.f);
  res.z = fmaxf((o.z - mean) * inv * g.z + bt.z, 0.f);
  res.w = fmaxf((o.w - mean) * inv * g.w + bt.w, 0.f);
  *(float4*)(out + (size_t)row * Dd + c) = res;
}

extern "C" void kernel_launch(void* const* d_in, const int* in_sizes, int n_in,
                              void* d_out, int out_size, void* d_ws, size_t ws_size,
                              hipStream_t stream) {
  const float* x       = (const float*)d_in[0];
  const int*   ei      = (const int*)  d_in[1];
  const float* W       = (const float*)d_in[2];
  const float* att_src = (const float*)d_in[3];
  const float* att_dst = (const float*)d_in[4];
  const float* bias    = (const float*)d_in[5];
  const float* gamma   = (const float*)d_in[6];
  const float* beta    = (const float*)d_in[7];
  float* out = (float*)d_out;

  // Workspace carve-out (~64 MB)
  char* p = (char*)d_ws;
  unsigned short* Wt = (unsigned short*)p; p += (((size_t)Dd * Dd * 2) + 255) & ~(size_t)255;
  float* h     = (float*)p; p += (size_t)Nn * Dd * 4;
  float* a_src = (float*)p; p += (size_t)Nn * Hh * 4;
  float* a_dst = (float*)p; p += (size_t)Nn * Hh * 4;
  float* m     = (float*)p; p += (size_t)Nn * Hh * 4;
  float* s     = (float*)p; p += (size_t)Nn * Hh * 4;

  k_init   <<<(Nn * Dd + 255) / 256, 256, 0, stream>>>(out, m, s);
  k_wcvt   <<<(Dd * Dd + 255) / 256, 256, 0, stream>>>(W, Wt);
  k_gemm   <<<(Nn + TM - 1) / TM,   256, 0, stream>>>(x, Wt, h);
  k_attn   <<<(Nn * Hh + 255) / 256, 256, 0, stream>>>(h, att_src, att_dst, a_src, a_dst);
  k_emax   <<<(Ee + 255) / 256,      256, 0, stream>>>(ei, a_src, a_dst, m);
  k_esum   <<<(Ee + 255) / 256,      256, 0, stream>>>(ei, a_src, a_dst, m, s);
  k_scatter<<<(Ee * 32 + 255) / 256, 256, 0, stream>>>(ei, a_src, a_dst, m, s, h, out);
  k_final  <<<(Nn * 32 + 255) / 256, 256, 0, stream>>>(x, bias, gamma, beta, out);
}
